// BallAttention_80204219285611
// MI455X (gfx1250) — matrix operand order repaired and checked by reading, not compiled, not run on hardware
//
#include <hip/hip_runtime.h>
#include <hip/hip_bf16.h>
#include <stdint.h>

#ifndef __has_builtin
#define __has_builtin(x) 0
#endif

typedef __bf16 bf16;
typedef __attribute__((ext_vector_type(16))) __bf16 v16bf;
typedef __attribute__((ext_vector_type(8)))  __bf16 v8bf;
typedef __attribute__((ext_vector_type(8)))  float  v8f;

#define N_TOK   131072      // B * n = 8 * 16384
#define DIM_    512
#define QKVN    1536        // 3 * DIM
#define NHEAD   8
#define HD_     64
#define GBALLS  2048        // B * n / 64
#define BALL_   64
#define PD_     3

// ---------------------------------------------------------------------------
// WMMA fragment loaders (CDNA5 §7.12.2 layouts, wave32)
// A (16x32 bf16): lane l holds row (l&15); vec elems 0..7 -> K = 8*(l>>4)+0..7,
//                 elems 8..15 -> K = 16 + 8*(l>>4) + 0..7
// B (32x16 bf16): lane l holds col (l&15); same K mapping as A
// C (16x16 f32):  elem r -> row r + 8*(l>>4), col (l&15)
// ---------------------------------------------------------------------------
static __device__ __forceinline__ v16bf frag_a(const bf16* row, int hf) {
  v8bf lo = *(const v8bf*)(row + 8 * hf);
  v8bf hi = *(const v8bf*)(row + 16 + 8 * hf);
  v16bf r;
#pragma unroll
  for (int e = 0; e < 8; ++e) { r[e] = lo[e]; r[e + 8] = hi[e]; }
  return r;
}
static __device__ __forceinline__ v16bf frag_b(const bf16* col, int hf) {
  v8bf lo = *(const v8bf*)(col + 8 * hf);
  v8bf hi = *(const v8bf*)(col + 16 + 8 * hf);
  v16bf r;
#pragma unroll
  for (int e = 0; e < 8; ++e) { r[e] = lo[e]; r[e + 8] = hi[e]; }
  return r;
}

// ---------------------------------------------------------------------------
// CDNA5 async global->LDS copy (ASYNCcnt path, ISA 08_async_tensor §4)
// ---------------------------------------------------------------------------
static __device__ __forceinline__ void async_b128(unsigned lds_off, const void* g) {
  asm volatile("global_load_async_to_lds_b128 %0, %1, off"
               :: "v"(lds_off), "v"((unsigned long long)(uintptr_t)g)
               : "memory");
}
template <int Nw>
static __device__ __forceinline__ void wait_async() {
#if __has_builtin(__builtin_amdgcn_s_wait_asynccnt)
  __builtin_amdgcn_s_wait_asynccnt(Nw);
#else
  if (Nw == 0) asm volatile("s_wait_asynccnt 0x0" ::: "memory");
  else         asm volatile("s_wait_asynccnt 0x1" ::: "memory");
#endif
}

// ---------------------------------------------------------------------------
// Kernel 1: per-ball mean of positions (G x 3)
// ---------------------------------------------------------------------------
__global__ __launch_bounds__(256) void ball_means_kernel(
    const float* __restrict__ pos, float* __restrict__ means) {
  int g = blockIdx.x * 256 + threadIdx.x;
  if (g >= GBALLS) return;
  float s0 = 0.f, s1 = 0.f, s2 = 0.f;
  const float* p = pos + (size_t)g * BALL_ * PD_;
#pragma unroll 4
  for (int i = 0; i < BALL_; ++i) {
    s0 += p[i * PD_ + 0]; s1 += p[i * PD_ + 1]; s2 += p[i * PD_ + 2];
  }
  means[g * PD_ + 0] = s0 * (1.f / BALL_);
  means[g * PD_ + 1] = s1 * (1.f / BALL_);
  means[g * PD_ + 2] = s2 * (1.f / BALL_);
}

// ---------------------------------------------------------------------------
// Kernel 2: x' = x + rel @ Wp + bp   (rel = pos - ball_mean), store bf16
// ---------------------------------------------------------------------------
__global__ __launch_bounds__(256) void xprime_kernel(
    const float* __restrict__ x, const float* __restrict__ pos,
    const float* __restrict__ means, const float* __restrict__ Wp,
    const float* __restrict__ bp, bf16* __restrict__ xp) {
  size_t id = (size_t)blockIdx.x * 256 + threadIdx.x;
  if (id >= (size_t)N_TOK * DIM_) return;
  int d = (int)(id & (DIM_ - 1));
  size_t t = id >> 9;              // token
  size_t g = t >> 6;               // ball
  float r0 = pos[t * PD_ + 0] - means[g * PD_ + 0];
  float r1 = pos[t * PD_ + 1] - means[g * PD_ + 1];
  float r2 = pos[t * PD_ + 2] - means[g * PD_ + 2];
  float v = x[id] + bp[d]
          + r0 * Wp[0 * DIM_ + d] + r1 * Wp[1 * DIM_ + d] + r2 * Wp[2 * DIM_ + d];
  xp[id] = (bf16)v;
}

// ---------------------------------------------------------------------------
// Kernel 3: fp32 (K x N row-major) -> bf16 (N x K) transpose
// ---------------------------------------------------------------------------
__global__ __launch_bounds__(256) void wt_kernel(
    const float* __restrict__ src, bf16* __restrict__ dst, int K, int N) {
  size_t id = (size_t)blockIdx.x * 256 + threadIdx.x;
  if (id >= (size_t)K * N) return;
  int n = (int)(id % N);
  int k = (int)(id / N);
  dst[(size_t)n * K + k] = (bf16)src[id];
}

// ---------------------------------------------------------------------------
// Kernel 4: WMMA GEMM  C[M,N] = A[M,K](bf16,row) @ Bt[N,K](bf16) + bias[N]
// Block: 256 thr = 8 waves. Block tile 256(M) x 64(N); wave -> 32 x 64
// (8 WMMA / k-step / wave). B tile (64x32 bf16 = 4KB) staged into LDS with
// CDNA5 async global->LDS loads, double buffered: one b128 async per thread.
// ---------------------------------------------------------------------------
template <bool OUT_BF16>
__global__ __launch_bounds__(256) void gemm_bf16_wmma(
    const bf16* __restrict__ A, const bf16* __restrict__ Bt,
    const float* __restrict__ bias, void* __restrict__ Cout,
    int M, int N, int K) {
  __shared__ __align__(16) bf16 sB[2][64][32];   // 2 x 4KB

  const int tid  = threadIdx.x;
  const int wave = tid >> 5;
  const int lane = tid & 31;
  const int hf   = lane >> 4;
  const int lm   = lane & 15;
  const int m0   = blockIdx.x * 256 + wave * 32;
  const int n0   = blockIdx.y * 64;

  // async B-stage: thread -> (col bc, 8-element part bp8) ; 256 thr * 16B = 4KB
  const int bc  = tid >> 2;
  const int bp8 = (tid & 3) * 8;
  const bf16* bsrc = Bt + (size_t)(n0 + bc) * K + bp8;
  const unsigned blds[2] = {
      (unsigned)(uintptr_t)&sB[0][bc][bp8],
      (unsigned)(uintptr_t)&sB[1][bc][bp8] };

  const bf16* arow0 = A + (size_t)(m0 + lm) * K;
  const bf16* arow1 = A + (size_t)(m0 + 16 + lm) * K;

  v8f acc[2][4] = {{v8f{}, v8f{}, v8f{}, v8f{}}, {v8f{}, v8f{}, v8f{}, v8f{}}};

  const int nk = K >> 5;           // k-steps of 32
  async_b128(blds[0], bsrc);       // prefetch k-step 0
  for (int ks = 0; ks < nk; ++ks) {
    const int k0 = ks << 5;
    if (ks + 1 < nk) {
      async_b128(blds[(ks + 1) & 1], bsrc + k0 + 32);
      wait_async<1>();             // own async for current buffer done (in order)
    } else {
      wait_async<0>();
    }
    __syncthreads();               // all waves' async data visible in LDS

    v16bf a0 = frag_a(arow0 + k0, hf);
    v16bf a1 = frag_a(arow1 + k0, hf);
    const bf16(*bb)[32] = sB[ks & 1];
#pragma unroll
    for (int t = 0; t < 4; ++t) {
      v16bf bfr = frag_b(&bb[t * 16 + lm][0], hf);
      acc[0][t] = __builtin_amdgcn_wmma_f32_16x16x32_bf16(
          false, a0, false, bfr, (short)0, acc[0][t], false, false);
      acc[1][t] = __builtin_amdgcn_wmma_f32_16x16x32_bf16(
          false, a1, false, bfr, (short)0, acc[1][t], false, false);
    }
    __syncthreads();               // done reading before next overwrite
  }

#pragma unroll
  for (int s = 0; s < 2; ++s) {
#pragma unroll
    for (int t = 0; t < 4; ++t) {
      int col = n0 + t * 16 + lm;
      float bv = bias[col];
#pragma unroll
      for (int r = 0; r < 8; ++r) {
        int row = m0 + s * 16 + r + 8 * hf;
        float v = acc[s][t][r] + bv;
        if (OUT_BF16) ((bf16*)Cout)[(size_t)row * N + col] = (bf16)v;
        else          ((float*)Cout)[(size_t)row * N + col] = v;
      }
    }
  }
}

// ---------------------------------------------------------------------------
// Kernel 5: per-(ball, head) attention. 128 thr = 4 waves; 64x64 tiles.
// qkv: bf16 [token][1536] = [q(8x64) | k(8x64) | v(8x64)]
// out: bf16 [token][512]  column = h*64 + d   (reference's final transpose)
// ---------------------------------------------------------------------------
__global__ __launch_bounds__(128) void ball_attn_kernel(
    const bf16* __restrict__ qkv, const float* __restrict__ pos,
    const float* __restrict__ sigma, bf16* __restrict__ aout) {
  const int g = blockIdx.x >> 3;
  const int h = blockIdx.x & 7;
  const int tid = threadIdx.x;
  const int wave = tid >> 5;
  const int lane = tid & 31;
  const int hf = lane >> 4;
  const int lm = lane & 15;

  __shared__ __align__(16) float spx[BALL_], spy[BALL_], spz[BALL_];
  __shared__ __align__(16) float sscore[BALL_][BALL_];   // 16 KB
  __shared__ __align__(16) bf16  sattn[BALL_][BALL_];    // 8 KB
  __shared__ __align__(16) bf16  svt[BALL_][BALL_];      // 8 KB, svt[d][j] = v[j][d]

  const bf16* qbase = qkv + (size_t)g * BALL_ * QKVN + h * HD_;
  const bf16* kbase = qbase + DIM_;
  const bf16* vbase = qbase + 2 * DIM_;

  for (int i = tid; i < BALL_; i += 128) {
    size_t t = (size_t)g * BALL_ + i;
    spx[i] = pos[t * PD_ + 0];
    spy[i] = pos[t * PD_ + 1];
    spz[i] = pos[t * PD_ + 2];
  }
  for (int idx = tid; idx < BALL_ * BALL_; idx += 128) {
    int j = idx >> 6, d = idx & 63;
    svt[d][j] = vbase[(size_t)j * QKVN + d];
  }
  __syncthreads();

  // ---- scores = q @ k^T * scale + sigma*dist ----
  v8f sc[4] = {v8f{}, v8f{}, v8f{}, v8f{}};
#pragma unroll
  for (int k0 = 0; k0 < HD_; k0 += 32) {
    v16bf af = frag_a(qbase + (size_t)(16 * wave + lm) * QKVN + k0, hf);
#pragma unroll
    for (int t = 0; t < 4; ++t) {
      v16bf bfr = frag_b(kbase + (size_t)(t * 16 + lm) * QKVN + k0, hf);
      sc[t] = __builtin_amdgcn_wmma_f32_16x16x32_bf16(
          false, af, false, bfr, (short)0, sc[t], false, false);
    }
  }
  const float sig = sigma[h];
  const float scale = 0.125f;  // 1/sqrt(64)
#pragma unroll
  for (int t = 0; t < 4; ++t) {
    int j = t * 16 + lm;
#pragma unroll
    for (int r = 0; r < 8; ++r) {
      int i = 16 * wave + r + 8 * hf;
      float dx = spx[i] - spx[j], dy = spy[i] - spy[j], dz = spz[i] - spz[j];
      float dist = sqrtf(dx * dx + dy * dy + dz * dz);
      sscore[i][j] = sc[t][r] * scale + sig * dist;
    }
  }
  __syncthreads();

  // ---- softmax over j, one row per thread (threads 0..63) ----
  if (tid < BALL_) {
    float mx = -3.4e38f;
#pragma unroll 4
    for (int j = 0; j < BALL_; ++j) mx = fmaxf(mx, sscore[tid][j]);
    float sum = 0.f;
#pragma unroll 4
    for (int j = 0; j < BALL_; ++j) {
      float e = __expf(sscore[tid][j] - mx);
      sscore[tid][j] = e;
      sum += e;
    }
    float inv = 1.f / sum;
#pragma unroll 4
    for (int j = 0; j < BALL_; ++j) sattn[tid][j] = (bf16)(sscore[tid][j] * inv);
  }
  __syncthreads();

  // ---- out = attn @ v ----
  v8f oc[4] = {v8f{}, v8f{}, v8f{}, v8f{}};
#pragma unroll
  for (int k0 = 0; k0 < BALL_; k0 += 32) {
    v16bf af = frag_a(&sattn[16 * wave + lm][k0], hf);
#pragma unroll
    for (int t = 0; t < 4; ++t) {
      v16bf bfr = frag_b(&svt[t * 16 + lm][k0], hf);
      oc[t] = __builtin_amdgcn_wmma_f32_16x16x32_bf16(
          false, af, false, bfr, (short)0, oc[t], false, false);
    }
  }
#pragma unroll
  for (int t = 0; t < 4; ++t) {
    int d = t * 16 + lm;
#pragma unroll
    for (int r = 0; r < 8; ++r) {
      int i = 16 * wave + r + 8 * hf;
      aout[((size_t)g * BALL_ + i) * DIM_ + h * HD_ + d] = (bf16)oc[t][r];
    }
  }
}

// ---------------------------------------------------------------------------
extern "C" void kernel_launch(void* const* d_in, const int* in_sizes, int n_in,
                              void* d_out, int out_size, void* d_ws, size_t ws_size,
                              hipStream_t stream) {
  (void)in_sizes; (void)n_in; (void)out_size; (void)ws_size;
  const float* x     = (const float*)d_in[0];
  const float* pos   = (const float*)d_in[1];
  const float* Wqkv  = (const float*)d_in[2];
  const float* bqkv  = (const float*)d_in[3];
  const float* Wp    = (const float*)d_in[4];
  const float* bp    = (const float*)d_in[5];
  const float* Wo    = (const float*)d_in[6];
  const float* bo    = (const float*)d_in[7];
  const float* sigma = (const float*)d_in[8];
  float* out = (float*)d_out;

  char* ws = (char*)d_ws;
  size_t off = 0;
  float* means = (float*)(ws + off); off += ((size_t)GBALLS * PD_ * 4 + 255) & ~(size_t)255;
  bf16* xprime = (bf16*)(ws + off);  off += (size_t)N_TOK * DIM_ * 2;
  bf16* wqkvT  = (bf16*)(ws + off);  off += (size_t)QKVN * DIM_ * 2;
  bf16* woT    = (bf16*)(ws + off);  off += (size_t)DIM_ * DIM_ * 2;
  bf16* qkvb   = (bf16*)(ws + off);  off += (size_t)N_TOK * QKVN * 2;
  bf16* aoutb  = (bf16*)(ws + off);  off += (size_t)N_TOK * DIM_ * 2;

  // 1. ball means
  ball_means_kernel<<<GBALLS / 256, 256, 0, stream>>>(pos, means);
  // 2. x' = x + rel@Wp + bp  (bf16)
  xprime_kernel<<<((size_t)N_TOK * DIM_) / 256, 256, 0, stream>>>(
      x, pos, means, Wp, bp, xprime);
  // 3. weights -> bf16 col-major
  wt_kernel<<<((size_t)DIM_ * QKVN + 255) / 256, 256, 0, stream>>>(Wqkv, wqkvT, DIM_, QKVN);
  wt_kernel<<<((size_t)DIM_ * DIM_ + 255) / 256, 256, 0, stream>>>(Wo, woT, DIM_, DIM_);
  // 4. QKV GEMM (WMMA + async LDS staging) -> bf16
  dim3 g1(N_TOK / 256, QKVN / 64);
  gemm_bf16_wmma<true><<<g1, 256, 0, stream>>>(xprime, wqkvT, bqkv, qkvb, N_TOK, QKVN, DIM_);
  // 5. ball attention (WMMA x2 + softmax + dist bias)
  ball_attn_kernel<<<GBALLS * NHEAD, 128, 0, stream>>>(qkvb, pos, sigma, aoutb);
  // 6. output projection (WMMA + async LDS staging) -> f32 d_out
  dim3 g2(N_TOK / 256, DIM_ / 64);
  gemm_bf16_wmma<false><<<g2, 256, 0, stream>>>(aoutb, woT, bo, out, N_TOK, DIM_, DIM_);
}
